// AVOD_79989470920835
// MI455X (gfx1250) — matrix operand order repaired and checked by reading, not compiled, not run on hardware
//
#include <hip/hip_runtime.h>
#include <hip/hip_fp16.h>

// ---------------- problem constants (match reference) ----------------
#define N_ANCH   4096
#define HIDDEN   256
#define ROI_SZ   3
#define RATIO    2
#define NSAMP    (ROI_SZ*RATIO)   // 6
#define IMG_H    360
#define IMG_W    1200
#define BEV_H    700
#define BEV_W    800
#define C_IN     32
#define NMS_THR  0.8f
#define NMS_SIZE 1024
#define X_MIN   (-40.0f)
#define Z_MIN    (0.0f)

typedef __attribute__((ext_vector_type(16))) _Float16 v16h;
typedef __attribute__((ext_vector_type(8)))  _Float16 v8h;
typedef __attribute__((ext_vector_type(8)))  float    v8f;

// =====================================================================
// 1) 1x1 conv (C_IN -> 1) + ReLU.  Memory bound: 127MB read total.
//    Coalesced across pixels; channel loop hits 32 planes sequentially.
// =====================================================================
__global__ void conv1x1_relu_kernel(const float* __restrict__ in,
                                    const float* __restrict__ w,
                                    const float* __restrict__ b,
                                    float* __restrict__ out, int hw) {
    int p = blockIdx.x * blockDim.x + threadIdx.x;
    if (p >= hw) return;
    float acc = b[0];
#pragma unroll
    for (int c = 0; c < C_IN; ++c)
        acc = fmaf(w[c], in[c * hw + p], acc);
    out[p] = fmaxf(acc, 0.0f);
}

// =====================================================================
// 2) RoIAlign (aligned=True, sampling_ratio=2, scale=1) on both maps,
//    fuse, write padded f16 activation matrix [N_ANCH][32] (cols 9..31=0)
//    so layer-1 can run as a single WMMA K-step.
// =====================================================================
__device__ __forceinline__ float bilinear(const float* __restrict__ f,
                                          int H, int W, float y, float x) {
    y = fminf(fmaxf(y, 0.0f), (float)(H - 1));
    x = fminf(fmaxf(x, 0.0f), (float)(W - 1));
    float y0f = floorf(y), x0f = floorf(x);
    int y0 = (int)y0f, x0 = (int)x0f;
    int y1 = y0 + 1 < H - 1 ? y0 + 1 : H - 1;
    int x1 = x0 + 1 < W - 1 ? x0 + 1 : W - 1;
    float ly = y - y0f, lx = x - x0f;
    float v00 = f[y0 * W + x0], v01 = f[y0 * W + x1];
    float v10 = f[y1 * W + x0], v11 = f[y1 * W + x1];
    return v00 * (1.f - ly) * (1.f - lx) + v01 * (1.f - ly) * lx +
           v10 * ly * (1.f - lx) + v11 * ly * lx;
}

__global__ void roi_fuse_kernel(const float* __restrict__ convImg,
                                const float* __restrict__ convBev,
                                const float* __restrict__ roisImg,
                                const float* __restrict__ roisBev,
                                const float* __restrict__ mImg,
                                const float* __restrict__ mBev,
                                _Float16* __restrict__ fused /*[N][32]*/) {
    int a = blockIdx.x * blockDim.x + threadIdx.x;
    if (a >= N_ANCH) return;
    float mi = mImg[0], mb = mBev[0];
    float bins[2][ROI_SZ * ROI_SZ];
    for (int m = 0; m < 2; ++m) {
        const float* feat = m ? convBev : convImg;
        const float* roi  = m ? roisBev : roisImg;
        int H = m ? BEV_H : IMG_H, W = m ? BEV_W : IMG_W;
        float x1 = roi[a * 5 + 1] - 0.5f;
        float y1 = roi[a * 5 + 2] - 0.5f;
        float rw = roi[a * 5 + 3] - roi[a * 5 + 1];
        float rh = roi[a * 5 + 4] - roi[a * 5 + 2];
        float acc[ROI_SZ * ROI_SZ];
#pragma unroll
        for (int i = 0; i < ROI_SZ * ROI_SZ; ++i) acc[i] = 0.f;
        for (int py = 0; py < NSAMP; ++py) {
            float y = y1 + rh * ((py + 0.5f) / NSAMP);
            for (int px = 0; px < NSAMP; ++px) {
                float x = x1 + rw * ((px + 0.5f) / NSAMP);
                acc[(py / RATIO) * ROI_SZ + (px / RATIO)] += bilinear(feat, H, W, y, x);
            }
        }
        float sc = (m ? mb : mi) * (1.0f / (RATIO * RATIO));
#pragma unroll
        for (int i = 0; i < ROI_SZ * ROI_SZ; ++i) bins[m][i] = acc[i] * sc;
    }
    float inv = 1.0f / (mi + mb);
#pragma unroll
    for (int j = 0; j < 32; ++j) {
        float v = (j < ROI_SZ * ROI_SZ) ? (bins[0][j] + bins[1][j]) * inv : 0.f;
        fused[a * 32 + j] = (_Float16)v;
    }
}

// =====================================================================
// 3) Weight prep: f32 -> f16, transposed to [N][K] row-major so B
//    fragments load as two contiguous 16B chunks per lane.
//    W1 [9,256] -> W1t f16 [256][32] (K padded to 32)
//    W2 [256,256] -> W2t f16 [256][256]
//    [W_obj|W_off] [256,8] -> Wot f16 [16][256] (N padded), bout f32[16]
// =====================================================================
__global__ void prep_weights_kernel(const float* __restrict__ W1,
                                    const float* __restrict__ W2,
                                    const float* __restrict__ Wobj,
                                    const float* __restrict__ Woff,
                                    const float* __restrict__ bobj,
                                    const float* __restrict__ boff,
                                    _Float16* __restrict__ W1t,
                                    _Float16* __restrict__ W2t,
                                    _Float16* __restrict__ Wot,
                                    float* __restrict__ bout) {
    int n = blockIdx.x * blockDim.x + threadIdx.x;
    if (n >= HIDDEN) return;
    for (int k = 0; k < 32; ++k)
        W1t[n * 32 + k] = (_Float16)((k < ROI_SZ * ROI_SZ) ? W1[k * HIDDEN + n] : 0.f);
    for (int k = 0; k < HIDDEN; ++k)
        W2t[n * HIDDEN + k] = (_Float16)W2[k * HIDDEN + n];
    if (n < 16) {
        for (int k = 0; k < HIDDEN; ++k) {
            float v = 0.f;
            if (n < 2)      v = Wobj[k * 2 + n];
            else if (n < 8) v = Woff[k * 6 + (n - 2)];
            Wot[n * HIDDEN + k] = (_Float16)v;
        }
        bout[n] = (n < 2) ? bobj[n] : (n < 8 ? boff[n - 2] : 0.f);
    }
}

// =====================================================================
// 4) WMMA GEMM: C[M,n_valid] = act(A[M,K] * Bt[N,K]^T + bias)
//    One wave per 16x16 output tile, K-stepped by 32 via
//    v_wmma_f32_16x16x32_f16 (f32 accumulate). Fragment loads use a
//    contiguous k order (k = 16h + s), same for A and B:
//      A: lane l(<16)=row l, elems K0..15; lane l+16 = row l, K16..31
//      B: lane l(<16)=col l, elems K0..15; lane l+16 = col l, K16..31
//    Tile guard is wave-uniform -> EXEC stays all-ones for WMMA.
// =====================================================================
union HFrag { v16h v; v8h h[2]; };

__global__ void gemm_wmma_kernel(const _Float16* __restrict__ A,   // [M][K]
                                 const _Float16* __restrict__ Bt,  // [N][K]
                                 const float* __restrict__ bias,   // [N]
                                 void* __restrict__ C,
                                 int M, int Nn, int K,
                                 int relu, int out_f16, int n_valid) {
    int wave = (int)((blockIdx.x * blockDim.x + threadIdx.x) >> 5);
    int lane = threadIdx.x & 31;
    int tiles_n = Nn >> 4;
    int tiles_m = M >> 4;
    if (wave >= tiles_m * tiles_n) return;      // wave-uniform
    int m0 = (wave / tiles_n) << 4;
    int n0 = (wave % tiles_n) << 4;
    int hs = lane >> 4;                         // half-wave select
    int r  = lane & 15;

    const _Float16* arow = A  + (size_t)(m0 + r) * K + hs * 16;
    const _Float16* brow = Bt + (size_t)(n0 + r) * K + hs * 16;

    v8f acc = {};
    for (int k = 0; k < K; k += 32) {
        HFrag a, b;
        a.h[0] = *(const v8h*)(arow + k);
        a.h[1] = *(const v8h*)(arow + k + 8);
        b.h[0] = *(const v8h*)(brow + k);
        b.h[1] = *(const v8h*)(brow + k + 8);
        acc = __builtin_amdgcn_wmma_f32_16x16x32_f16(
            false, a.v, false, b.v, (short)0, acc, false, false);
    }

    int col = n0 + r;
    float bcol = bias[col];
#pragma unroll
    for (int i = 0; i < 8; ++i) {
        int row = m0 + i + hs * 8;              // C/D layout: vgpr i, half hs
        float v = acc[i] + bcol;
        if (relu) v = fmaxf(v, 0.0f);
        if (col < n_valid) {
            if (out_f16) ((_Float16*)C)[(size_t)row * n_valid + col] = (_Float16)v;
            else         ((float*)C)[(size_t)row * n_valid + col] = v;
        }
    }
}

// =====================================================================
// 5) Per-anchor post-processing: offsets->anchors, BEV projection,
//    stable softmax score.
// =====================================================================
__global__ void postproc_kernel(const float* __restrict__ logits, // [N][8]
                                const float* __restrict__ fanch,  // [N][6]
                                float* __restrict__ reg,          // [N][6]
                                float* __restrict__ obj,          // [N][2]
                                float* __restrict__ bevb,         // [N][4]
                                float* __restrict__ score) {      // [N]
    int a = blockIdx.x * blockDim.x + threadIdx.x;
    if (a >= N_ANCH) return;
    const float* l  = logits + (size_t)a * 8;
    const float* fa = fanch  + (size_t)a * 6;
    float rg[6];
#pragma unroll
    for (int i = 0; i < 3; ++i) rg[i]     = fa[i] + l[2 + i] * fa[3 + i];
#pragma unroll
    for (int i = 0; i < 3; ++i) rg[3 + i] = fa[3 + i] * expf(l[5 + i]);
#pragma unroll
    for (int i = 0; i < 6; ++i) reg[a * 6 + i] = rg[i];
    float bx = rg[0] - X_MIN, bz = rg[2] - Z_MIN;
    bevb[a * 4 + 0] = bx - rg[3] * 0.5f;
    bevb[a * 4 + 1] = bz - rg[5] * 0.5f;
    bevb[a * 4 + 2] = bx + rg[3] * 0.5f;
    bevb[a * 4 + 3] = bz + rg[5] * 0.5f;
    float o0 = l[0], o1 = l[1];
    obj[a * 2 + 0] = o0; obj[a * 2 + 1] = o1;
    float m = fmaxf(o0, o1);
    float e0 = expf(o0 - m), e1 = expf(o1 - m);
    score[a] = e1 / (e0 + e1);
}

// =====================================================================
// 6) Bitonic sort of 4096 (score desc, index asc tie-break) in LDS.
//    Single workgroup; 32KB static LDS.
// =====================================================================
__global__ void sort_scores_kernel(const float* __restrict__ score,
                                   int* __restrict__ order) {
    __shared__ float s[N_ANCH];
    __shared__ int   id[N_ANCH];
    int t = threadIdx.x, nt = blockDim.x;
    for (int i = t; i < N_ANCH; i += nt) { s[i] = score[i]; id[i] = i; }
    __syncthreads();
    for (int k = 2; k <= N_ANCH; k <<= 1) {
        for (int j = k >> 1; j > 0; j >>= 1) {
            for (int idx = t; idx < N_ANCH; idx += nt) {
                int ixj = idx ^ j;
                if (ixj > idx) {
                    bool dir = ((idx & k) == 0);
                    float si = s[idx], sj = s[ixj];
                    int   ii = id[idx], ij = id[ixj];
                    // "before": idx-element correctly precedes ixj-element
                    bool before = (si > sj) || (si == sj && ii < ij);
                    bool sw = dir ? !before : before;
                    if (sw) { s[idx] = sj; s[ixj] = si; id[idx] = ij; id[ixj] = ii; }
                }
            }
            __syncthreads();
        }
    }
    for (int i = t; i < N_ANCH; i += nt) order[i] = id[i];
}

// =====================================================================
// 7) Greedy NMS on sorted boxes (LDS-resident, 96KB dynamic LDS — fits
//    easily in CDNA5's 320KB/WGP), stable-partition selection of top
//    NMS_SIZE, gather final [1024][8] output.
// =====================================================================
__global__ void nms_select_kernel(const int* __restrict__ order,
                                  const float* __restrict__ bevb,
                                  const float* __restrict__ reg,
                                  const float* __restrict__ obj,
                                  float* __restrict__ out) {
    extern __shared__ char smem[];
    float* bx1 = (float*)smem;
    float* bz1 = bx1 + N_ANCH;
    float* bx2 = bz1 + N_ANCH;
    float* bz2 = bx2 + N_ANCH;
    int*  keep = (int*)(bz2 + N_ANCH);
    int*  pos  = keep + N_ANCH;
    int t = threadIdx.x, nt = blockDim.x;
    for (int i = t; i < N_ANCH; i += nt) {
        int o = order[i];
        bx1[i] = bevb[o * 4 + 0]; bz1[i] = bevb[o * 4 + 1];
        bx2[i] = bevb[o * 4 + 2]; bz2[i] = bevb[o * 4 + 3];
        keep[i] = 1;
    }
    __syncthreads();
    for (int i = 0; i < N_ANCH - 1; ++i) {
        if (keep[i]) {
            float ax1 = bx1[i], az1 = bz1[i], ax2 = bx2[i], az2 = bz2[i];
            float aarea = (ax2 - ax1) * (az2 - az1);
            for (int j = i + 1 + t; j < N_ANCH; j += nt) {
                if (keep[j]) {
                    float ix1 = fmaxf(ax1, bx1[j]), iz1 = fmaxf(az1, bz1[j]);
                    float ix2 = fminf(ax2, bx2[j]), iz2 = fminf(az2, bz2[j]);
                    float iw = fmaxf(ix2 - ix1, 0.f), ih = fmaxf(iz2 - iz1, 0.f);
                    float inter = iw * ih;
                    float barea = (bx2[j] - bx1[j]) * (bz2[j] - bz1[j]);
                    float iou = inter / (aarea + barea - inter + 1e-8f);
                    if (iou > NMS_THR) keep[j] = 0;
                }
            }
        }
        __syncthreads();
    }
    // stable partition: kept first (score order), then suppressed
    if (t == 0) {
        int ck = 0;
        for (int i = 0; i < N_ANCH; ++i) if (keep[i])  { pos[i] = ck; ck++; }
        int cu = 0;
        for (int i = 0; i < N_ANCH; ++i) if (!keep[i]) { pos[i] = ck + cu; cu++; }
    }
    __syncthreads();
    for (int i = t; i < N_ANCH; i += nt) {
        int p = pos[i];
        if (p < NMS_SIZE) {
            int o = order[i];
#pragma unroll
            for (int c = 0; c < 6; ++c) out[p * 8 + c] = reg[o * 6 + c];
            out[p * 8 + 6] = obj[o * 2 + 0];
            out[p * 8 + 7] = obj[o * 2 + 1];
        }
    }
}

// =====================================================================
// Launcher
// =====================================================================
static inline size_t align256(size_t x) { return (x + 255) & ~(size_t)255; }

extern "C" void kernel_launch(void* const* d_in, const int* in_sizes, int n_in,
                              void* d_out, int out_size, void* d_ws, size_t ws_size,
                              hipStream_t stream) {
    (void)in_sizes; (void)n_in; (void)out_size; (void)ws_size;
    // input order per setup_inputs()
    const float* img_map  = (const float*)d_in[0];
    const float* bev_map  = (const float*)d_in[1];
    const float* anch_img = (const float*)d_in[2];
    const float* anch_bev = (const float*)d_in[3];
    const float* fanch    = (const float*)d_in[4];
    const float* img_mask = (const float*)d_in[5];
    const float* bev_mask = (const float*)d_in[6];
    const float* w_img    = (const float*)d_in[7];
    const float* b_img    = (const float*)d_in[8];
    const float* w_bev    = (const float*)d_in[9];
    const float* b_bev    = (const float*)d_in[10];
    const float* W1       = (const float*)d_in[11];
    const float* b1       = (const float*)d_in[12];
    const float* W2       = (const float*)d_in[13];
    const float* b2       = (const float*)d_in[14];
    const float* W_obj    = (const float*)d_in[15];
    const float* b_obj    = (const float*)d_in[16];
    const float* W_off    = (const float*)d_in[17];
    const float* b_off    = (const float*)d_in[18];
    float* out = (float*)d_out;

    // workspace layout
    char* ws = (char*)d_ws;
    size_t off = 0;
    const int IMG_HW = IMG_H * IMG_W, BEV_HW = BEV_H * BEV_W;
    float* convImg = (float*)(ws + off); off = align256(off + (size_t)IMG_HW * 4);
    float* convBev = (float*)(ws + off); off = align256(off + (size_t)BEV_HW * 4);
    _Float16* fused = (_Float16*)(ws + off); off = align256(off + (size_t)N_ANCH * 32 * 2);
    _Float16* W1t   = (_Float16*)(ws + off); off = align256(off + (size_t)HIDDEN * 32 * 2);
    _Float16* W2t   = (_Float16*)(ws + off); off = align256(off + (size_t)HIDDEN * HIDDEN * 2);
    _Float16* Wot   = (_Float16*)(ws + off); off = align256(off + (size_t)16 * HIDDEN * 2);
    float* bout     = (float*)(ws + off);    off = align256(off + 16 * 4);
    _Float16* h1    = (_Float16*)(ws + off); off = align256(off + (size_t)N_ANCH * HIDDEN * 2);
    _Float16* h2    = (_Float16*)(ws + off); off = align256(off + (size_t)N_ANCH * HIDDEN * 2);
    float* logits   = (float*)(ws + off);    off = align256(off + (size_t)N_ANCH * 8 * 4);
    float* reg      = (float*)(ws + off);    off = align256(off + (size_t)N_ANCH * 6 * 4);
    float* obj      = (float*)(ws + off);    off = align256(off + (size_t)N_ANCH * 2 * 4);
    float* bevb     = (float*)(ws + off);    off = align256(off + (size_t)N_ANCH * 4 * 4);
    float* score    = (float*)(ws + off);    off = align256(off + (size_t)N_ANCH * 4);
    int* order      = (int*)(ws + off);      off = align256(off + (size_t)N_ANCH * 4);

    // 1) memory-bound 1x1 convs
    conv1x1_relu_kernel<<<(IMG_HW + 255) / 256, 256, 0, stream>>>(img_map, w_img, b_img, convImg, IMG_HW);
    conv1x1_relu_kernel<<<(BEV_HW + 255) / 256, 256, 0, stream>>>(bev_map, w_bev, b_bev, convBev, BEV_HW);

    // 2) weight prep (tiny)
    prep_weights_kernel<<<1, HIDDEN, 0, stream>>>(W1, W2, W_obj, W_off, b_obj, b_off,
                                                  W1t, W2t, Wot, bout);

    // 3) RoIAlign + fuse -> padded f16 activations
    roi_fuse_kernel<<<N_ANCH / 256, 256, 0, stream>>>(convImg, convBev, anch_img, anch_bev,
                                                      img_mask, bev_mask, fused);

    // 4) MLP via WMMA. tiles = (M/16)*(N/16) waves, 8 waves/block.
    {
        int tiles = (N_ANCH / 16) * (HIDDEN / 16);         // 4096 waves
        int blocks = (tiles * 32 + 255) / 256;
        // layer 1: [4096,32] x [32,256] -> h1 (relu, f16)
        gemm_wmma_kernel<<<blocks, 256, 0, stream>>>(fused, W1t, b1, h1,
                                                     N_ANCH, HIDDEN, 32, 1, 1, HIDDEN);
        // layer 2: [4096,256] x [256,256] -> h2 (relu, f16)
        gemm_wmma_kernel<<<blocks, 256, 0, stream>>>(h1, W2t, b2, h2,
                                                     N_ANCH, HIDDEN, HIDDEN, 1, 1, HIDDEN);
        // layer 3: [4096,256] x [256,16pad] -> logits f32 [4096][8]
        int tiles3 = (N_ANCH / 16) * (16 / 16);            // 256 waves
        int blocks3 = (tiles3 * 32 + 255) / 256;
        gemm_wmma_kernel<<<blocks3, 256, 0, stream>>>(h2, Wot, bout, logits,
                                                      N_ANCH, 16, HIDDEN, 0, 0, 8);
    }

    // 5) per-anchor decode
    postproc_kernel<<<N_ANCH / 256, 256, 0, stream>>>(logits, fanch, reg, obj, bevb, score);

    // 6) sort by score (desc) — single WG bitonic in LDS
    sort_scores_kernel<<<1, 1024, 0, stream>>>(score, order);

    // 7) NMS + select + gather output [1024][8]
    size_t nms_lds = (size_t)N_ANCH * 4 * sizeof(float) + (size_t)N_ANCH * 2 * sizeof(int); // 96KB
    nms_select_kernel<<<1, 1024, nms_lds, stream>>>(order, bevb, reg, obj, out);
}
